// QLstmNetwork_3367254360720
// MI455X (gfx1250) — compile-verified, matrix operand order checked by reading, not run on hardware
//
#include <hip/hip_runtime.h>
#include <hip/hip_bf16.h>

// ---------------------------------------------------------------------------
// MI455X (gfx1250) QLstmNetwork forward.
// bf16 WMMA LSTM scan, batch-tiled (32 WGs x 16 rows), c resident in VGPRs,
// h recirculated through LDS.
// Round 6: round-5's uintptr_t laundering (which fixed the LICM spill) also
// erased address-space provenance -> all weight loads became flat_load_b128
// (FLAT uses LOADcnt *and* DScnt, coupling the weight stream to the LDS
// counter: s_wait_loadcnt_dscnt 0x101/0x202). Fix: reconstruct the pointer
// as an explicit address_space(1) (global) pointer from the laundered
// integer, restoring global_load_b128 with independent LOADcnt/DScnt waits.
// ---------------------------------------------------------------------------

typedef __attribute__((ext_vector_type(16))) __bf16       v16bf;
typedef __attribute__((ext_vector_type(8)))  float        v8f;
typedef __attribute__((ext_vector_type(4)))  unsigned int v4u;

// global (address_space(1)) pointer to 16-byte chunks
typedef const v4u __attribute__((address_space(1)))* gv4p;

union Frag { v4u u[2]; v16bf v; };

#define BATCH  512
#define TT     256
#define FF     128
#define HH     1024
#define FEATP  2080          // FEAT=2050 padded to multiple of 32
#define HALFW  512

// packed fragment counts (each fragment = 32x16 bf16 = 512 elements = 1 KB)
#define LSTM_FRAGS   (4 * 64 * 36)            // gates * hidden-blocks * k-tiles
#define LSTM_ELEMS   (LSTM_FRAGS * 512)       // 4,718,592
#define HEAD_FRAGS   (4 * 32 * 65)            // matrices * n-tiles * k-tiles
#define HEAD_ELEMS   (HEAD_FRAGS * 512)       // 4,259,840
#define FEAT_ELEMS   (BATCH * FEATP)          // 1,064,960

#define GATE_STRIDE  ((size_t)64 * 36 * 1024) // bytes between gate packs (2,359,296)

__device__ __forceinline__ float sigmoid_f(float x) {
  return 1.0f / (1.0f + __expf(-x));
}
__device__ __forceinline__ float tanh_f(float x) {
  x = fminf(fmaxf(x, -15.0f), 15.0f);
  float e = __expf(2.0f * x);
  return (e - 1.0f) / (e + 1.0f);
}

// Load a 32-byte B fragment through an explicit GLOBAL address-space pointer
// (emits global_load_b128, LOADcnt only — even when the base was laundered).
__device__ __forceinline__ Frag load_frag_gas(uintptr_t p) {
  Frag f;
  gv4p q = (gv4p)p;
  f.u[0] = q[0];
  f.u[1] = q[1];
  return f;
}
__device__ __forceinline__ Frag load_frag_g(const char* p) {
  Frag f;
  f.u[0] = *(const v4u*)p;
  f.u[1] = *(const v4u*)(p + 16);
  return f;
}
// LDS A-fragment: lanes 0-15 take K(+0..7,+16..23), lanes 16-31 K(+8..15,+24..31)
__device__ __forceinline__ Frag load_frag_a(const __bf16* base, int hsel) {
  Frag f;
  const v4u* ap = (const v4u*)((const char*)base + hsel * 16);
  f.u[0] = ap[0];
  f.u[1] = ap[2];   // +32 bytes
  return f;
}

// ---------------------------------------------------------------------------
// Pack W_lstm (1152x4096 f32 row-major) into bf16 WMMA-B fragments.
// B-frag layout (mirror of ISA 16-bit A layout): lane<16: N=lane, vgpr v<4
// holds K = 2v,2v+1; v>=4 holds K = 16+2(v-4)..; lane>=16: +8 on K.
// Lane-major storage: lane owns 32 contiguous bytes -> 2x b128 per fragment.
// ---------------------------------------------------------------------------
__global__ void pack_lstm_kernel(const float* __restrict__ W,
                                 __bf16* __restrict__ out, int total) {
  int idx = blockIdx.x * blockDim.x + threadIdx.x;
  if (idx >= total) return;
  int e    = idx & 15;
  int lane = (idx >> 4) & 31;
  int frag = idx >> 9;
  int kt   = frag % 36;
  int rest = frag / 36;
  int hb   = rest & 63;
  int g    = rest >> 6;
  int v = e >> 1, lo = e & 1, hsel = lane >> 4;
  int koff = (v < 4 ? v * 2 + lo : 16 + (v - 4) * 2 + lo) + hsel * 8;
  int K = kt * 32 + koff;
  int N = g * 1024 + hb * 16 + (lane & 15);
  out[idx] = (__bf16)W[(size_t)K * 4096 + N];
}

__global__ void pack_heads_kernel(const float* __restrict__ W3,
                                  const float* __restrict__ W4,
                                  const float* __restrict__ W5,
                                  const float* __restrict__ W6,
                                  __bf16* __restrict__ out, int total) {
  int idx = blockIdx.x * blockDim.x + threadIdx.x;
  if (idx >= total) return;
  int e    = idx & 15;
  int lane = (idx >> 4) & 31;
  int frag = idx >> 9;
  int kt   = frag % 65;
  int rest = frag / 65;
  int nt   = rest & 31;
  int m    = rest >> 5;
  int v = e >> 1, lo = e & 1, hsel = lane >> 4;
  int koff = (v < 4 ? v * 2 + lo : 16 + (v - 4) * 2 + lo) + hsel * 8;
  int K = kt * 32 + koff;
  int N = nt * 16 + (lane & 15);
  const float* W = (m == 0) ? W3 : (m == 1) ? W4 : (m == 2) ? W5 : W6;
  float val = (K < 2050) ? W[(size_t)K * HALFW + N] : 0.0f;
  out[idx] = (__bf16)val;
}

__global__ void zero_loss_kernel(float* out) {
  if (threadIdx.x == 0 && blockIdx.x == 0) out[0] = 0.0f;
}

// ---------------------------------------------------------------------------
// LSTM scan. grid=32 (B/16), block=512 (16 waves). Wave w owns hidden blocks
// hb = 4w..4w+3. Two barriers per timestep.
// ---------------------------------------------------------------------------
__global__ __launch_bounds__(512, 1)
void lstm_scan_kernel(const float* __restrict__ hist,
                      const float* __restrict__ head,
                      const float* __restrict__ b_lstm,
                      const __bf16* __restrict__ wpk,
                      __bf16* __restrict__ featb) {
  __shared__ alignas(16) __bf16 xbuf[16][FF];
  __shared__ alignas(16) __bf16 hbuf[16][HH];

  const int tid    = threadIdx.x;
  const int wave   = tid >> 5;
  const int lane   = tid & 31;
  const int hsel   = lane >> 4;     // 0: lanes 0-15, 1: lanes 16-31
  const int mcol   = lane & 15;
  const int wgbase = blockIdx.x * 16;

  // zero the recurrent h buffer
  {
    __bf16* hz = &hbuf[0][0];
    for (int i = tid; i < 16 * HH; i += 512) hz[i] = (__bf16)0.0f;
  }

  const v8f zero8 = {0.f, 0.f, 0.f, 0.f, 0.f, 0.f, 0.f, 0.f};
  v8f creg[4] = {zero8, zero8, zero8, zero8};
  v8f hreg[4] = {zero8, zero8, zero8, zero8};

  // per-wave gate biases (zeros in setup, kept for generality)
  float bias_i[4], bias_j[4], bias_f[4], bias_o[4];
#pragma unroll
  for (int blk = 0; blk < 4; ++blk) {
    const int hb = (wave << 2) + blk;
    bias_i[blk] = b_lstm[0 * 1024 + hb * 16 + mcol];
    bias_j[blk] = b_lstm[1 * 1024 + hb * 16 + mcol];
    bias_f[blk] = b_lstm[2 * 1024 + hb * 16 + mcol];
    bias_o[blk] = b_lstm[3 * 1024 + hb * 16 + mcol];
  }

#pragma unroll 1
  for (int t = 0; t < TT; ++t) {
    // stage x_t (16 rows x 128 cols): row = wave, cols = lane*4..+3
    {
      const float4 xv = *(const float4*)&hist[((size_t)(wgbase + wave) * TT + t) * FF + lane * 4];
      xbuf[wave][lane * 4 + 0] = (__bf16)xv.x;
      xbuf[wave][lane * 4 + 1] = (__bf16)xv.y;
      xbuf[wave][lane * 4 + 2] = (__bf16)xv.z;
      xbuf[wave][lane * 4 + 3] = (__bf16)xv.w;
    }
    __syncthreads();   // x ready; h writes of previous step visible

#pragma unroll
    for (int blk = 0; blk < 4; ++blk) {
      // Opaque B base: defeats LICM of the kt=0 peel's weight loads out of
      // the t loop (round-4 scratch spill). Loads below go through an
      // explicit address_space(1) pointer so they stay global_load_b128.
      uintptr_t bb = (uintptr_t)((const char*)wpk +
          (((size_t)((wave << 2) + blk) * 36) << 10) + ((size_t)lane << 5));
      asm volatile("" : "+v"(bb));

      v8f acc0, acc1, acc2, acc3;

      // ---- kt = 0 (xbuf), C = inline zero: defines the accumulators ----
      {
        Frag a  = load_frag_a(&xbuf[mcol][0], hsel);
        Frag b0 = load_frag_gas(bb + 0 * GATE_STRIDE);
        Frag b1 = load_frag_gas(bb + 1 * GATE_STRIDE);
        Frag b2 = load_frag_gas(bb + 2 * GATE_STRIDE);
        Frag b3 = load_frag_gas(bb + 3 * GATE_STRIDE);
        acc0 = __builtin_amdgcn_wmma_f32_16x16x32_bf16(false, a.v, false, b0.v, (short)0, zero8, false, false);
        acc1 = __builtin_amdgcn_wmma_f32_16x16x32_bf16(false, a.v, false, b1.v, (short)0, zero8, false, false);
        acc2 = __builtin_amdgcn_wmma_f32_16x16x32_bf16(false, a.v, false, b2.v, (short)0, zero8, false, false);
        acc3 = __builtin_amdgcn_wmma_f32_16x16x32_bf16(false, a.v, false, b3.v, (short)0, zero8, false, false);
      }
      // ---- kt = 1..35, rolled (no spills, 4 B-frags in flight) ----
#pragma unroll 1
      for (int kt = 1; kt < 36; ++kt) {
        const int kb = kt * 32;
        const __bf16* abase = (kb < FF) ? &xbuf[mcol][kb] : &hbuf[mcol][kb - FF];
        Frag a = load_frag_a(abase, hsel);
        const uintptr_t p = bb + (size_t)kt * 1024;
        Frag b0 = load_frag_gas(p + 0 * GATE_STRIDE);
        Frag b1 = load_frag_gas(p + 1 * GATE_STRIDE);
        Frag b2 = load_frag_gas(p + 2 * GATE_STRIDE);
        Frag b3 = load_frag_gas(p + 3 * GATE_STRIDE);
        acc0 = __builtin_amdgcn_wmma_f32_16x16x32_bf16(false, a.v, false, b0.v, (short)0, acc0, false, false);
        acc1 = __builtin_amdgcn_wmma_f32_16x16x32_bf16(false, a.v, false, b1.v, (short)0, acc1, false, false);
        acc2 = __builtin_amdgcn_wmma_f32_16x16x32_bf16(false, a.v, false, b2.v, (short)0, acc2, false, false);
        acc3 = __builtin_amdgcn_wmma_f32_16x16x32_bf16(false, a.v, false, b3.v, (short)0, acc3, false, false);
      }

      // gate math in C/D register layout (identical across the 4 gate tiles)
#pragma unroll
      for (int r = 0; r < 8; ++r) {
        float iv = acc0[r] + bias_i[blk];
        float jv = acc1[r] + bias_j[blk];
        float fv = acc2[r] + bias_f[blk];
        float ov = acc3[r] + bias_o[blk];
        float cn = sigmoid_f(fv + 1.0f) * creg[blk][r] + sigmoid_f(iv) * tanh_f(jv);
        creg[blk][r] = cn;
        hreg[blk][r] = sigmoid_f(ov) * tanh_f(cn);
      }
    }
    __syncthreads();   // all reads of hbuf done -> safe to overwrite

#pragma unroll
    for (int blk = 0; blk < 4; ++blk) {
      const int col = ((wave << 2) + blk) * 16 + mcol;
#pragma unroll
      for (int r = 0; r < 8; ++r)
        hbuf[r + hsel * 8][col] = (__bf16)hreg[blk][r];
    }
  }

  // epilogue: feature = [pnl(2) | c(1024) | h(1024) | pad(30)] as bf16
#pragma unroll
  for (int blk = 0; blk < 4; ++blk) {
    const int col = ((wave << 2) + blk) * 16 + mcol;
#pragma unroll
    for (int r = 0; r < 8; ++r) {
      const int row = r + hsel * 8;
      const size_t fb = (size_t)(wgbase + row) * FEATP;
      featb[fb + 2 + col]      = (__bf16)creg[blk][r];
      featb[fb + 2 + HH + col] = (__bf16)hreg[blk][r];
    }
  }
  if (tid < 16) {
    const int bb = wgbase + tid;
    const size_t fb = (size_t)bb * FEATP;
    featb[fb + 0] = (__bf16)head[bb * 3 + 1];
    featb[fb + 1] = (__bf16)head[bb * 3 + 2];
    for (int k = 2 + 2 * HH; k < FEATP; ++k) featb[fb + k] = (__bf16)0.0f;
  }
}

// ---------------------------------------------------------------------------
// Heads: 4 GEMMs (16 x 2080 x 512) via WMMA, relu, tiny duel layers via LDS
// float atomics, mask/argmax/MSE. grid=32, block=512.
// jobid = wave + 16*j, j=0..7 -> (m = jobid/32, nt = jobid%32).
// ---------------------------------------------------------------------------
__global__ __launch_bounds__(512, 1)
void heads_kernel(const __bf16* __restrict__ featb,
                  const __bf16* __restrict__ wpkh,
                  const float* __restrict__ b3, const float* __restrict__ b4,
                  const float* __restrict__ b5, const float* __restrict__ b6,
                  const float* __restrict__ W_sell_v, const float* __restrict__ b_sell_v,
                  const float* __restrict__ W_buy_v,  const float* __restrict__ b_buy_v,
                  const float* __restrict__ W_sell_h, const float* __restrict__ b_sell_h,
                  const float* __restrict__ W_buy_h,  const float* __restrict__ b_buy_h,
                  const float* __restrict__ head,
                  const float* __restrict__ targetQ,
                  const int* __restrict__ action,
                  float* __restrict__ out) {
  __shared__ float s_pv[16], s_ev[16];
  __shared__ float s_ph[16][2], s_eh[16][2];

  const int tid    = threadIdx.x;
  const int wave   = tid >> 5;
  const int lane   = tid & 31;
  const int hsel   = lane >> 4;
  const int mcol   = lane & 15;
  const int wgbase = blockIdx.x * 16;

  if (tid < 16) {
    s_pv[tid] = 0.f; s_ev[tid] = 0.f;
    s_ph[tid][0] = 0.f; s_ph[tid][1] = 0.f;
    s_eh[tid][0] = 0.f; s_eh[tid][1] = 0.f;
  }
  __syncthreads();

  const v8f zero8 = {0.f, 0.f, 0.f, 0.f, 0.f, 0.f, 0.f, 0.f};
  const int brow = wgbase + mcol;     // A-operand batch row for this lane
  const char* abase = (const char*)(featb + (size_t)brow * FEATP) + hsel * 16;

#pragma unroll 1
  for (int j = 0; j < 8; ++j) {
    const int jobid = wave + (j << 4);
    const int m  = jobid >> 5;        // which of W3/W4/W5/W6
    const int nt = jobid & 31;        // 16-col tile of the 512 outputs
    const char* bb = (const char*)wpkh + (((size_t)(m * 32 + nt) * 65) << 10) + ((size_t)lane << 5);

    // kt = 0 peeled with C = inline zero
    Frag a0, f0;
    a0.u[0] = *(const v4u*)abase;
    a0.u[1] = *(const v4u*)(abase + 32);
    f0 = load_frag_g(bb);
    v8f acc = __builtin_amdgcn_wmma_f32_16x16x32_bf16(false, a0.v, false, f0.v, (short)0, zero8, false, false);

#pragma unroll 1
    for (int kt = 1; kt < 65; ++kt) {
      Frag a, bfrag;
      const char* ap = abase + (size_t)kt * 64;
      a.u[0] = *(const v4u*)ap;
      a.u[1] = *(const v4u*)(ap + 32);
      bfrag = load_frag_g(bb + (size_t)kt * 1024);
      acc = __builtin_amdgcn_wmma_f32_16x16x32_bf16(
          false, a.v, false, bfrag.v, (short)0, acc, false, false);
    }

    const float* bp2 = (m == 0) ? b3 : (m == 1) ? b4 : (m == 2) ? b5 : b6;
    const float bias = bp2[nt * 16 + mcol];
    const int k = nt * 16 + mcol;
#pragma unroll
    for (int r = 0; r < 8; ++r) {
      const float val = fmaxf(acc[r] + bias, 0.0f);
      const int row = r + hsel * 8;
      if (m == 0) {
        atomicAdd(&s_pv[row], val * W_sell_v[k]);
      } else if (m == 1) {
        atomicAdd(&s_ph[row][0], val * W_sell_h[k * 2 + 0]);
        atomicAdd(&s_ph[row][1], val * W_sell_h[k * 2 + 1]);
      } else if (m == 2) {
        atomicAdd(&s_ev[row], val * W_buy_v[k]);
      } else {
        atomicAdd(&s_eh[row][0], val * W_buy_h[k * 2 + 0]);
        atomicAdd(&s_eh[row][1], val * W_buy_h[k * 2 + 1]);
      }
    }
  }
  __syncthreads();

  if (tid < 16) {
    const int row = tid;
    const int bb = wgbase + row;
    float pv  = s_pv[row] + b_sell_v[0];
    float pa0 = s_ph[row][0] + b_sell_h[0];
    float pa1 = s_ph[row][1] + b_sell_h[1];
    float pme = 0.5f * (pa0 + pa1);
    float pQ0 = pa0 - pme + pv, pQ1 = pa1 - pme + pv;
    float ev  = s_ev[row] + b_buy_v[0];
    float ea0 = s_eh[row][0] + b_buy_h[0];
    float ea1 = s_eh[row][1] + b_buy_h[1];
    float eme = 0.5f * (ea0 + ea1);
    float eQ0 = ea0 - eme + ev, eQ1 = ea1 - eme + ev;

    float isp = head[bb * 3 + 0];
    float Q0 = (isp != 0.0f) ? pQ0 : eQ0;            // mask col0 = is_pos
    float Q1 = ((1.0f - isp) != 0.0f) ? pQ1 : eQ1;   // mask col1 = 1-is_pos
    int greedy = (Q1 > Q0) ? 1 : 0;
    float infv = (action[bb] == 0) ? Q0 : Q1;
    float d = targetQ[bb] - infv;
    atomicAdd(&out[0], d * d * (1.0f / (float)BATCH));
    out[1 + bb * 2 + 0] = Q0;
    out[1 + bb * 2 + 1] = Q1;
    ((int*)out)[1 + 2 * BATCH + bb] = greedy;
  }
}

// ---------------------------------------------------------------------------
extern "C" void kernel_launch(void* const* d_in, const int* in_sizes, int n_in,
                              void* d_out, int out_size, void* d_ws, size_t ws_size,
                              hipStream_t stream) {
  const float* hist     = (const float*)d_in[0];
  const float* head     = (const float*)d_in[1];
  const float* targetQ  = (const float*)d_in[2];
  const int*   action   = (const int*)d_in[3];
  const float* W_lstm   = (const float*)d_in[4];
  const float* b_lstm   = (const float*)d_in[5];
  const float* W3       = (const float*)d_in[6];
  const float* b3       = (const float*)d_in[7];
  const float* W4       = (const float*)d_in[8];
  const float* b4       = (const float*)d_in[9];
  const float* W5       = (const float*)d_in[10];
  const float* b5       = (const float*)d_in[11];
  const float* W6       = (const float*)d_in[12];
  const float* b6       = (const float*)d_in[13];
  const float* W_sell_v = (const float*)d_in[14];
  const float* b_sell_v = (const float*)d_in[15];
  const float* W_buy_v  = (const float*)d_in[16];
  const float* b_buy_v  = (const float*)d_in[17];
  const float* W_sell_h = (const float*)d_in[18];
  const float* b_sell_h = (const float*)d_in[19];
  const float* W_buy_h  = (const float*)d_in[20];
  const float* b_buy_h  = (const float*)d_in[21];

  // workspace layout (bf16, all sub-offsets 16B aligned)
  __bf16* wpk   = (__bf16*)d_ws;            // LSTM_ELEMS
  __bf16* wpkh  = wpk + LSTM_ELEMS;         // HEAD_ELEMS
  __bf16* featb = wpkh + HEAD_ELEMS;        // FEAT_ELEMS

  zero_loss_kernel<<<1, 32, 0, stream>>>((float*)d_out);
  pack_lstm_kernel<<<(LSTM_ELEMS + 255) / 256, 256, 0, stream>>>(W_lstm, wpk, LSTM_ELEMS);
  pack_heads_kernel<<<(HEAD_ELEMS + 255) / 256, 256, 0, stream>>>(W3, W4, W5, W6, wpkh, HEAD_ELEMS);
  lstm_scan_kernel<<<BATCH / 16, 512, 0, stream>>>(hist, head, b_lstm, wpk, featb);
  heads_kernel<<<BATCH / 16, 512, 0, stream>>>(featb, wpkh, b3, b4, b5, b6,
                                               W_sell_v, b_sell_v, W_buy_v, b_buy_v,
                                               W_sell_h, b_sell_h, W_buy_h, b_buy_h,
                                               head, targetQ, action, (float*)d_out);
}
